// Block_14602888806424
// MI455X (gfx1250) — matrix operand order repaired and checked by reading, not compiled, not run on hardware
//
#include <hip/hip_runtime.h>
#include <hip/hip_bf16.h>
#include <math.h>

#define BB   2
#define TT   2048
#define CC   2048
#define HH   32
#define NNh  64
#define FF   7168
#define BT   (BB*TT)
#define DMIX 32
#define DDEC 64
#define NMIX (5*DMIX)   /* 160 */

typedef __hip_bfloat16 bf16_t;
typedef __attribute__((ext_vector_type(16))) __bf16 v16bf;
typedef __attribute__((ext_vector_type(8)))  float  v8f;

// ---- CDNA5 async global->LDS copy (ASYNCcnt-tracked, bypasses VGPRs) ----
__device__ __forceinline__ void async_b128(unsigned lds_off, const bf16_t* g) {
  asm volatile("global_load_async_to_lds_b128 %0, %1, off"
               :: "v"(lds_off), "v"(g)
               : "memory");
}
__device__ __forceinline__ void wait_async0() {
  asm volatile("s_wait_asynccnt 0x0" ::: "memory");
}

// ---------------------------------------------------------------- cvt + transpose
// dst[n*K + k] = bf16(src[k*N + n])  (weights pre-transposed so GEMM B-tiles
// stage as contiguous 16B chunks -> async b128 loads, no in-LDS transpose)
__global__ __launch_bounds__(256) void cvt_transpose_kernel(
    const float* __restrict__ src, bf16_t* __restrict__ dst, int K, int N) {
  __shared__ float tile[32][33];
  int kb = blockIdx.y * 32, nb = blockIdx.x * 32;
  int tx = threadIdx.x & 31, ty = threadIdx.x >> 5;   // 32 x 8
#pragma unroll
  for (int i = 0; i < 32; i += 8)
    tile[ty + i][tx] = src[(size_t)(kb + ty + i) * N + nb + tx];
  __syncthreads();
#pragma unroll
  for (int i = 0; i < 32; i += 8)
    dst[(size_t)(nb + ty + i) * K + kb + tx] = __float2bfloat16(tile[tx][ty + i]);
}

// ---------------------------------------------------------------- per-token LayerNorm
__global__ __launch_bounds__(256) void ln_kernel(const float* __restrict__ x,
                                                 const float* __restrict__ w,
                                                 const float* __restrict__ b,
                                                 float* __restrict__ out) {
  int row = blockIdx.x;
  const float* xr = x + (size_t)row * CC;
  float vals[8];
  float s = 0.f;
#pragma unroll
  for (int i = 0; i < 8; i++) { vals[i] = xr[threadIdx.x + i * 256]; s += vals[i]; }
  __shared__ float red[256];
  red[threadIdx.x] = s; __syncthreads();
  for (int off = 128; off > 0; off >>= 1) {
    if ((int)threadIdx.x < off) red[threadIdx.x] += red[threadIdx.x + off];
    __syncthreads();
  }
  float mean = red[0] * (1.f / CC);
  __syncthreads();
  float s2 = 0.f;
#pragma unroll
  for (int i = 0; i < 8; i++) { float d = vals[i] - mean; s2 += d * d; }
  red[threadIdx.x] = s2; __syncthreads();
  for (int off = 128; off > 0; off >>= 1) {
    if ((int)threadIdx.x < off) red[threadIdx.x] += red[threadIdx.x + off];
    __syncthreads();
  }
  float rstd = rsqrtf(red[0] * (1.f / CC) + 1e-5f);
  float* orow = out + (size_t)row * CC;
#pragma unroll
  for (int i = 0; i < 8; i++) {
    int c = threadIdx.x + i * 256;
    orow[c] = (vals[i] - mean) * rstd * w[c] + b[c];
  }
}

// ---------------------------------------------------------------- fused token mixing MLP
__global__ __launch_bounds__(256) void token_mlp_kernel(
    const float* __restrict__ h,
    const float* __restrict__ tm_maa, const float* __restrict__ maa_w1,
    const float* __restrict__ maa_w2, const float* __restrict__ time_decay,
    const float* __restrict__ td_w1,  const float* __restrict__ td_w2,
    bf16_t* __restrict__ xk_bf, bf16_t* __restrict__ xv_bf,
    bf16_t* __restrict__ xr_bf, bf16_t* __restrict__ xg_bf,
    float* __restrict__ dec) {
  int row = blockIdx.x;
  int t   = row & (TT - 1);
  int tid = threadIdx.x;
  __shared__ float h_s[CC];
  __shared__ float xx_s[CC];
  __shared__ float xw_s[CC];     // first holds xxx, later holds xw
  __shared__ float a_s[NMIX];
  __shared__ float td_s[DDEC];

  const float* hr = h + (size_t)row * CC;
  const float* hp = hr - CC;
#pragma unroll
  for (int i = 0; i < 8; i++) {
    int c = tid + i * 256;
    float hv = hr[c];
    float pv = (t > 0) ? hp[c] : 0.f;
    h_s[c] = hv;
    xx_s[c] = pv - hv;
  }
  __syncthreads();
#pragma unroll
  for (int i = 0; i < 8; i++) {
    int c = tid + i * 256;
    xw_s[c] = h_s[c] + xx_s[c] * tm_maa[c];   // xxx
  }
  __syncthreads();
  if (tid < NMIX) {
    float acc = 0.f;
    for (int c = 0; c < CC; c++) acc += xw_s[c] * maa_w1[(size_t)c * NMIX + tid];
    a_s[tid] = tanhf(acc);
  }
  __syncthreads();
#pragma unroll
  for (int i = 0; i < 8; i++) {
    int c = tid + i * 256;
    float hv = h_s[c], xxv = xx_s[c];
    float m0 = 0.f, m1 = 0.f, m2 = 0.f, m3 = 0.f, m4 = 0.f;
    for (int d = 0; d < DMIX; d++) {
      m0 += a_s[0 * DMIX + d] * maa_w2[(size_t)(0 * DMIX + d) * CC + c];
      m1 += a_s[1 * DMIX + d] * maa_w2[(size_t)(1 * DMIX + d) * CC + c];
      m2 += a_s[2 * DMIX + d] * maa_w2[(size_t)(2 * DMIX + d) * CC + c];
      m3 += a_s[3 * DMIX + d] * maa_w2[(size_t)(3 * DMIX + d) * CC + c];
      m4 += a_s[4 * DMIX + d] * maa_w2[(size_t)(4 * DMIX + d) * CC + c];
    }
    size_t o = (size_t)row * CC + c;
    xw_s[c]  = hv + xxv * (tm_maa[1 * CC + c] + m0);
    xk_bf[o] = __float2bfloat16(hv + xxv * (tm_maa[2 * CC + c] + m1));
    xv_bf[o] = __float2bfloat16(hv + xxv * (tm_maa[3 * CC + c] + m2));
    xr_bf[o] = __float2bfloat16(hv + xxv * (tm_maa[4 * CC + c] + m3));
    xg_bf[o] = __float2bfloat16(hv + xxv * (tm_maa[5 * CC + c] + m4));
  }
  __syncthreads();
  if (tid < DDEC) {
    float acc = 0.f;
    for (int c = 0; c < CC; c++) acc += xw_s[c] * td_w1[(size_t)c * DDEC + tid];
    td_s[tid] = tanhf(acc);
  }
  __syncthreads();
#pragma unroll
  for (int i = 0; i < 8; i++) {
    int c = tid + i * 256;
    float acc = time_decay[c];
    for (int d = 0; d < DDEC; d++) acc += td_s[d] * td_w2[(size_t)d * CC + c];
    dec[(size_t)row * CC + c] = __expf(-__expf(acc));
  }
}

// ---------------------------------------------------------------- bf16 WMMA GEMM
// out(MxN) = A(MxK, bf16 row-major) @ B, where B is supplied TRANSPOSED:
// Bt is [N][K] bf16 row-major (weights pre-transposed by cvt_transpose_kernel).
// Tiling: 128x128 block tile, BK=32, 256 threads = 8 waves (4x2 grid), each
// wave 2x4 fragments of v_wmma_f32_16x16x32_bf16. Both tiles staged into
// double-buffered LDS with GLOBAL_LOAD_ASYNC_TO_LDS_B128 (ASYNCcnt), so
// staging of tile t+1 overlaps WMMA on tile t; one barrier per K-step.
// Epilogue modes: 0=none 1=silu 2=relu^2->bf16 3=sigmoid 4=res+acc 5=res+scale*acc
__global__ __launch_bounds__(256) void gemm_bf16_kernel(
    const bf16_t* __restrict__ A, const bf16_t* __restrict__ Bt,
    int M, int N, int K,
    const float* __restrict__ res, const float* __restrict__ scale,
    float* __restrict__ out_f, bf16_t* __restrict__ out_h, int mode) {
  __shared__ __align__(16) bf16_t As[2][128][32];
  __shared__ __align__(16) bf16_t Bs[2][128][32];   // [n][k]
  int tid  = threadIdx.x;
  int lane = tid & 31;
  int wid  = tid >> 5;
  int wr   = wid >> 1;       // 0..3 : 32-row strip
  int wc   = wid & 1;        // 0..1 : 64-col strip
  int lanlo = lane & 15;
  int lanhi = lane >> 4;
  int m_blk = blockIdx.y * 128;
  int n_blk = blockIdx.x * 128;

  // staging coordinates: 512 x 16B chunks per tile, 2 per thread per matrix
  int r0 = tid >> 2;               // 0..63
  int j0 = (tid & 3) * 8;          // 0,8,16,24
  const bf16_t* Ab = A  + (size_t)m_blk * K;
  const bf16_t* Bb = Bt + (size_t)n_blk * K;

  v8f acc[2][4] = {};

  // prologue: stage tile 0 into buffer 0
#pragma unroll
  for (int rep = 0; rep < 2; rep++) {
    int rr = r0 + rep * 64;
    async_b128((unsigned)(size_t)&As[0][rr][j0], Ab + (size_t)rr * K + j0);
    async_b128((unsigned)(size_t)&Bs[0][rr][j0], Bb + (size_t)rr * K + j0);
  }

  int nk = K >> 5;
  for (int t = 0; t < nk; t++) {
    int buf = t & 1;
    wait_async0();        // my async writes into As[buf]/Bs[buf] done
    __syncthreads();      // everyone's done (each waited before arriving)
    if (t + 1 < nk) {
      int k0 = (t + 1) << 5;
#pragma unroll
      for (int rep = 0; rep < 2; rep++) {
        int rr = r0 + rep * 64;
        async_b128((unsigned)(size_t)&As[buf ^ 1][rr][j0], Ab + (size_t)rr * K + k0 + j0);
        async_b128((unsigned)(size_t)&Bs[buf ^ 1][rr][j0], Bb + (size_t)rr * K + k0 + j0);
      }
      if (t + 2 < nk) {   // light prefetch two tiles ahead (global_prefetch_b8)
        __builtin_prefetch(Ab + (size_t)r0 * K + ((t + 2) << 5), 0, 1);
        __builtin_prefetch(Bb + (size_t)r0 * K + ((t + 2) << 5), 0, 1);
      }
    }
    // ---- compute: 8 WMMAs per K step, fragments straight from LDS
#pragma unroll
    for (int fr = 0; fr < 2; fr++) {
      int arow = wr * 32 + fr * 16 + lanlo;
      union { uint4 u[2]; v16bf v; } af;
      af.u[0] = *((const uint4*)&As[buf][arow][lanhi * 8]);
      af.u[1] = *((const uint4*)&As[buf][arow][16 + lanhi * 8]);
#pragma unroll
      for (int fc = 0; fc < 4; fc++) {
        int bcol = wc * 64 + fc * 16 + lanlo;
        union { uint4 u[2]; v16bf v; } bfm;
        bfm.u[0] = *((const uint4*)&Bs[buf][bcol][lanhi * 8]);
        bfm.u[1] = *((const uint4*)&Bs[buf][bcol][16 + lanhi * 8]);
        acc[fr][fc] = __builtin_amdgcn_wmma_f32_16x16x32_bf16(
            false, af.v, false, bfm.v, (short)0, acc[fr][fc], false, false);
      }
    }
  }

  // ---- epilogue (C/D layout: VGPR g -> M = g + 8*lanhi, N = lanlo)
#pragma unroll
  for (int fr = 0; fr < 2; fr++) {
#pragma unroll
    for (int fc = 0; fc < 4; fc++) {
#pragma unroll
      for (int g8 = 0; g8 < 8; g8++) {
        int m = m_blk + wr * 32 + fr * 16 + g8 + lanhi * 8;
        int n = n_blk + wc * 64 + fc * 16 + lanlo;
        size_t o = (size_t)m * N + n;
        float v = acc[fr][fc][g8];
        if (mode == 0)      out_f[o] = v;
        else if (mode == 1) out_f[o] = v / (1.f + __expf(-v));                 // silu
        else if (mode == 2) { float rv = v > 0.f ? v : 0.f; out_h[o] = __float2bfloat16(rv * rv); }
        else if (mode == 3) out_f[o] = 1.f / (1.f + __expf(-v));               // sigmoid
        else if (mode == 4) out_f[o] = res[o] + v;                             // residual
        else                out_f[o] = res[o] + scale[o] * v;                  // res + s*acc
      }
    }
  }
}

// ---------------------------------------------------------------- WKV6 recurrent scan
__global__ __launch_bounds__(64) void wkv_scan_kernel(
    const float* __restrict__ r, const float* __restrict__ k,
    const float* __restrict__ v, const float* __restrict__ dec,
    const float* __restrict__ u, float* __restrict__ y) {
  int bh = blockIdx.x;
  int b  = bh / HH;
  int hh = bh % HH;
  int j  = threadIdx.x;
  __shared__ float S[NNh][NNh + 1];
  __shared__ float rs[NNh], ks[NNh], ds_[NNh], us[NNh];
  for (int i = 0; i < NNh; i++) S[i][j] = 0.f;
  us[j] = u[hh * NNh + j];
  __syncthreads();
  size_t base = (size_t)b * TT * CC + (size_t)hh * NNh;
  for (int t = 0; t < TT; t++) {
    size_t off = base + (size_t)t * CC;
    rs[j] = r[off + j]; ks[j] = k[off + j]; ds_[j] = dec[off + j];
    float vj = v[off + j];
    __syncthreads();
    float s1 = 0.f, yj = 0.f;
#pragma unroll 8
    for (int i = 0; i < NNh; i++) {
      float ri = rs[i], ki = ks[i];
      s1 += ri * us[i] * ki;
      float Sij = S[i][j];
      yj += ri * Sij;
      S[i][j] = Sij * ds_[i] + ki * vj;
    }
    y[off + j] = yj + s1 * vj;
    __syncthreads();
  }
}

// ---------------------------------------------------------------- GroupNorm(y)*g -> bf16
__global__ __launch_bounds__(256) void gn_mul_kernel(
    const float* __restrict__ y, const float* __restrict__ lw,
    const float* __restrict__ lb, const float* __restrict__ g,
    bf16_t* __restrict__ out) {
  int row = blockIdx.x;
  int tid = threadIdx.x;
  const float* yr = y + (size_t)row * CC;
  int c0 = tid * 8;          // 8 contiguous channels, head = tid>>3
  float vals[8];
  float s = 0.f;
#pragma unroll
  for (int i = 0; i < 8; i++) { vals[i] = yr[c0 + i]; s += vals[i]; }
  __shared__ float red[256];
  __shared__ float hstat[HH];
  red[tid] = s; __syncthreads();
  if ((tid & 7) == 0) {
    float a = 0.f;
    for (int jj = 0; jj < 8; jj++) a += red[tid + jj];
    hstat[tid >> 3] = a * (1.f / NNh);
  }
  __syncthreads();
  float mean = hstat[tid >> 3];
  __syncthreads();
  float s2 = 0.f;
#pragma unroll
  for (int i = 0; i < 8; i++) { float d = vals[i] - mean; s2 += d * d; }
  red[tid] = s2; __syncthreads();
  if ((tid & 7) == 0) {
    float a = 0.f;
    for (int jj = 0; jj < 8; jj++) a += red[tid + jj];
    hstat[tid >> 3] = rsqrtf(a * (1.f / NNh) + 6.4e-4f);   // eps = 1e-5*8^2
  }
  __syncthreads();
  float rstd = hstat[tid >> 3];
  const float* gr = g + (size_t)row * CC;
  bf16_t* orow = out + (size_t)row * CC;
#pragma unroll
  for (int i = 0; i < 8; i++) {
    int c = c0 + i;
    float v = (vals[i] - mean) * rstd * lw[c] + lb[c];
    orow[c] = __float2bfloat16(v * gr[c]);
  }
}

// ---------------------------------------------------------------- channel-mix shift+mix
__global__ __launch_bounds__(256) void shift_mix2_kernel(
    const float* __restrict__ h, const float* __restrict__ cm,
    bf16_t* __restrict__ o0, bf16_t* __restrict__ o1) {
  int row = blockIdx.x;
  int t   = row & (TT - 1);
  int tid = threadIdx.x;
  const float* hr = h + (size_t)row * CC;
  const float* hp = hr - CC;
#pragma unroll
  for (int i = 0; i < 8; i++) {
    int c = tid + i * 256;
    float hv  = hr[c];
    float xxv = ((t > 0) ? hp[c] : 0.f) - hv;
    size_t o = (size_t)row * CC + c;
    o0[o] = __float2bfloat16(hv + xxv * cm[c]);        // k-mix
    o1[o] = __float2bfloat16(hv + xxv * cm[CC + c]);   // r-mix
  }
}

// ================================================================ host orchestration
extern "C" void kernel_launch(void* const* d_in, const int* in_sizes, int n_in,
                              void* d_out, int out_size, void* d_ws, size_t ws_size,
                              hipStream_t stream) {
  (void)in_sizes; (void)n_in; (void)out_size; (void)ws_size;
  const float* x          = (const float*)d_in[0];
  const float* ln1_w      = (const float*)d_in[1];
  const float* ln1_b      = (const float*)d_in[2];
  const float* ln2_w      = (const float*)d_in[3];
  const float* ln2_b      = (const float*)d_in[4];
  const float* tm_maa     = (const float*)d_in[5];
  const float* maa_w1     = (const float*)d_in[6];
  const float* maa_w2     = (const float*)d_in[7];
  const float* time_decay = (const float*)d_in[8];
  const float* td_w1      = (const float*)d_in[9];
  const float* td_w2      = (const float*)d_in[10];
  const float* time_faaaa = (const float*)d_in[11];
  const float* Wr         = (const float*)d_in[12];
  const float* Wk         = (const float*)d_in[13];
  const float* Wv         = (const float*)d_in[14];
  const float* Wg         = (const float*)d_in[15];
  const float* Wo         = (const float*)d_in[16];
  const float* lnx_w      = (const float*)d_in[17];
  const float* lnx_b      = (const float*)d_in[18];
  const float* cm_maa     = (const float*)d_in[19];
  const float* Wck        = (const float*)d_in[20];
  const float* Wcr        = (const float*)d_in[21];
  const float* Wcv        = (const float*)d_in[22];

  char* wsp = (char*)d_ws;
  auto alloc = [&](size_t bytes) -> void* {
    void* p = (void*)wsp;
    wsp += (bytes + 255) & ~(size_t)255;
    return p;
  };
  size_t actF = (size_t)BT * CC * sizeof(float);
  size_t actH = (size_t)BT * CC * sizeof(bf16_t);
  float* h_    = (float*)alloc(actF);
  float* r_    = (float*)alloc(actF);
  float* k_    = (float*)alloc(actF);
  float* v_    = (float*)alloc(actF);
  float* g_    = (float*)alloc(actF);
  float* dec_  = (float*)alloc(actF);
  float* y_    = (float*)alloc(actF);
  float* x2_   = (float*)alloc(actF);
  float* sig_  = (float*)alloc(actF);
  bf16_t* xk_bf = (bf16_t*)alloc(actH);
  bf16_t* xv_bf = (bf16_t*)alloc(actH);
  bf16_t* xr_bf = (bf16_t*)alloc(actH);
  bf16_t* xg_bf = (bf16_t*)alloc(actH);
  bf16_t* gn_bf = (bf16_t*)alloc(actH);
  bf16_t* kf_bf = (bf16_t*)alloc((size_t)BT * FF * sizeof(bf16_t));
  // transposed bf16 weights: Wt[n][k]
  bf16_t* WrT  = (bf16_t*)alloc((size_t)CC * CC * sizeof(bf16_t));
  bf16_t* WkT  = (bf16_t*)alloc((size_t)CC * CC * sizeof(bf16_t));
  bf16_t* WvT  = (bf16_t*)alloc((size_t)CC * CC * sizeof(bf16_t));
  bf16_t* WgT  = (bf16_t*)alloc((size_t)CC * CC * sizeof(bf16_t));
  bf16_t* WoT  = (bf16_t*)alloc((size_t)CC * CC * sizeof(bf16_t));
  bf16_t* WcrT = (bf16_t*)alloc((size_t)CC * CC * sizeof(bf16_t));
  bf16_t* WckT = (bf16_t*)alloc((size_t)CC * FF * sizeof(bf16_t));  // [FF][CC]
  bf16_t* WcvT = (bf16_t*)alloc((size_t)FF * CC * sizeof(bf16_t));  // [CC][FF]

  auto cvtT = [&](const float* s, bf16_t* d, int K, int N) {
    cvt_transpose_kernel<<<dim3(N / 32, K / 32), dim3(256), 0, stream>>>(s, d, K, N);
  };
  cvtT(Wr,  WrT,  CC, CC);
  cvtT(Wk,  WkT,  CC, CC);
  cvtT(Wv,  WvT,  CC, CC);
  cvtT(Wg,  WgT,  CC, CC);
  cvtT(Wo,  WoT,  CC, CC);
  cvtT(Wcr, WcrT, CC, CC);
  cvtT(Wck, WckT, CC, FF);
  cvtT(Wcv, WcvT, FF, CC);

  auto gemm = [&](const bf16_t* Aa, const bf16_t* BtT, int M, int N, int K,
                  const float* res, const float* sc, float* of, bf16_t* oh, int mode) {
    gemm_bf16_kernel<<<dim3(N / 128, M / 128), dim3(256), 0, stream>>>(
        Aa, BtT, M, N, K, res, sc, of, oh, mode);
  };

  // ---- time mixing ----
  ln_kernel<<<dim3(BT), dim3(256), 0, stream>>>(x, ln1_w, ln1_b, h_);
  token_mlp_kernel<<<dim3(BT), dim3(256), 0, stream>>>(
      h_, tm_maa, maa_w1, maa_w2, time_decay, td_w1, td_w2,
      xk_bf, xv_bf, xr_bf, xg_bf, dec_);
  gemm(xr_bf, WrT, BT, CC, CC, nullptr, nullptr, r_, nullptr, 0);
  gemm(xk_bf, WkT, BT, CC, CC, nullptr, nullptr, k_, nullptr, 0);
  gemm(xv_bf, WvT, BT, CC, CC, nullptr, nullptr, v_, nullptr, 0);
  gemm(xg_bf, WgT, BT, CC, CC, nullptr, nullptr, g_, nullptr, 1);   // silu
  wkv_scan_kernel<<<dim3(BB * HH), dim3(64), 0, stream>>>(r_, k_, v_, dec_, time_faaaa, y_);
  gn_mul_kernel<<<dim3(BT), dim3(256), 0, stream>>>(y_, lnx_w, lnx_b, g_, gn_bf);
  gemm(gn_bf, WoT, BT, CC, CC, x, nullptr, x2_, nullptr, 4);        // x2 = x + att

  // ---- channel mixing ----
  ln_kernel<<<dim3(BT), dim3(256), 0, stream>>>(x2_, ln2_w, ln2_b, h_);
  shift_mix2_kernel<<<dim3(BT), dim3(256), 0, stream>>>(h_, cm_maa, xk_bf, xr_bf);
  gemm(xr_bf, WcrT, BT, CC, CC, nullptr, nullptr, sig_, nullptr, 3);   // sigmoid
  gemm(xk_bf, WckT, BT, FF, CC, nullptr, nullptr, nullptr, kf_bf, 2);  // relu^2 -> bf16
  gemm(kf_bf, WcvT, BT, CC, FF, x2_, sig_, (float*)d_out, nullptr, 5); // out = x2 + sig*acc
}
